// GATBlock_87342454931667
// MI455X (gfx1250) — matrix operand order repaired and checked by reading, not compiled, not run on hardware
//
#include <hip/hip_runtime.h>
#include <hip/hip_bf16.h>

// ---------------------------------------------------------------------------
// CDNA5 (gfx1250) GAT-VAE implementation.
// Heavy GEMMs use v_wmma_f32_16x16x32_bf16 (wave32 WMMA), everything else is
// f32 VALU / global atomics. All launches on `stream`, graph-capture safe.
// ---------------------------------------------------------------------------

typedef __attribute__((ext_vector_type(16))) __bf16 v16bf;
typedef __attribute__((ext_vector_type(8)))  float  v8f;

union FragBF {
    v16bf v;
    uint4 q[2];
};

__device__ __forceinline__ unsigned short f32_to_bf16_rne(float f) {
    union { float f; unsigned u; } c;
    c.f = f;
    unsigned u = c.u;
    unsigned r = u + 0x7FFFu + ((u >> 16) & 1u);
    // NaN guard
    if ((u & 0x7F800000u) == 0x7F800000u) r = u;
    return (unsigned short)(r >> 16);
}

// ---------------------------------------------------------------------------
// Weight prep: transpose f32 [K x Nn] -> bf16 column-major [Nn x Kpad], 0-pad K
// ---------------------------------------------------------------------------
__global__ __launch_bounds__(256) void w_transpose_bf16(
    const float* __restrict__ W, unsigned short* __restrict__ out,
    int K, int Nn, int Kpad)
{
    int idx = blockIdx.x * 256 + threadIdx.x;
    if (idx >= Nn * Kpad) return;
    int n = idx / Kpad;
    int k = idx - n * Kpad;
    out[idx] = (k < K) ? f32_to_bf16_rne(W[(size_t)k * Nn + n]) : (unsigned short)0;
}

__global__ __launch_bounds__(256) void w_convert_bf16(
    const float* __restrict__ W, unsigned short* __restrict__ out, int n)
{
    int i = blockIdx.x * 256 + threadIdx.x;
    if (i < n) out[i] = f32_to_bf16_rne(W[i]);
}

__global__ __launch_bounds__(256) void fill_u32(
    unsigned* __restrict__ p, unsigned v, int n)
{
    int i = blockIdx.x * 256 + threadIdx.x;
    if (i < n) p[i] = v;
}

// ---------------------------------------------------------------------------
// WMMA GEMM:  C[M x Nout] = A[M x K] * Bt^T   (Bt stored column-major [Nout x ldb])
// A_BF16: 0 -> A is f32 (converted in-register via v_cvt_pk_bf16_f32,
//              K tail peeled out of the hot loop)
//         1 -> A is bf16 (K must be a multiple of 32)
// ACT:    0 -> none, 1 -> softplus
// Wave computes a 32x64 tile: 2 A-frags x 4 B-frags -> 8 wmma per k-step
// (16 b128 loads / 8 wmma = 2.0). Block = 8 waves covering 256x64 of C.
// ---------------------------------------------------------------------------
template <int A_BF16, int ACT>
__global__ __launch_bounds__(256) void gemm_wmma(
    const void* __restrict__ Aptr, const unsigned short* __restrict__ Bt,
    float* __restrict__ C, int M, int K, int Nout, int lda, int ldb, int ldc)
{
    const int lane = threadIdx.x & 31;
    const int wave = threadIdx.x >> 5;
    const int m0 = blockIdx.y * 256 + wave * 32;
    const int n0 = blockIdx.x * 64;

    int rowA0 = m0 + (lane & 15);
    int rowA1 = m0 + 16 + (lane & 15);
    if (rowA0 >= M) rowA0 = M - 1;               // clamp; stores are guarded
    if (rowA1 >= M) rowA1 = M - 1;
    const int kA = (lane >> 4) * 8;              // A frag: el0-7=k+kA.., el8-15=k+kA+16..
    const int kB = (lane >> 4) * 8;              // B frag: el0-7=k+kB.., el8-15=k+kB+16..

    const unsigned short* brow[4];
#pragma unroll
    for (int j = 0; j < 4; ++j) {
        int c = n0 + j * 16 + (lane & 15);
        if (c >= Nout) c = Nout - 1;
        brow[j] = Bt + (size_t)c * ldb;
    }

    v8f acc[8];
#pragma unroll
    for (int j = 0; j < 8; ++j) acc[j] = (v8f){0.f,0.f,0.f,0.f,0.f,0.f,0.f,0.f};

    auto loadA = [&](int row, int k0, FragBF& a) {
        if (A_BF16) {
            const unsigned short* ap =
                (const unsigned short*)Aptr + (size_t)row * lda + k0 + kA;
            a.q[0] = *(const uint4*)(ap);
            a.q[1] = *(const uint4*)(ap + 16);
        } else {
            const float* ap = (const float*)Aptr + (size_t)row * lda + k0 + kA;
            __builtin_prefetch(ap + 64, 0, 1);   // k-tile two steps ahead of this row
            float4 f0 = *(const float4*)(ap);
            float4 f1 = *(const float4*)(ap + 4);
            float4 f2 = *(const float4*)(ap + 16);
            float4 f3 = *(const float4*)(ap + 20);
            a.v[0]=(__bf16)f0.x; a.v[1]=(__bf16)f0.y; a.v[2]=(__bf16)f0.z; a.v[3]=(__bf16)f0.w;
            a.v[4]=(__bf16)f1.x; a.v[5]=(__bf16)f1.y; a.v[6]=(__bf16)f1.z; a.v[7]=(__bf16)f1.w;
            a.v[8]=(__bf16)f2.x; a.v[9]=(__bf16)f2.y; a.v[10]=(__bf16)f2.z; a.v[11]=(__bf16)f2.w;
            a.v[12]=(__bf16)f3.x; a.v[13]=(__bf16)f3.y; a.v[14]=(__bf16)f3.z; a.v[15]=(__bf16)f3.w;
        }
    };
    auto loadB = [&](int k0, FragBF (&b)[4]) {
#pragma unroll
        for (int j = 0; j < 4; ++j) {
            const unsigned short* bp = brow[j] + k0 + kB;
            b[j].q[0] = *(const uint4*)(bp);
            b[j].q[1] = *(const uint4*)(bp + 16);
        }
    };
    auto mmaStep = [&](const FragBF& a0, const FragBF& a1, const FragBF (&b)[4]) {
#pragma unroll
        for (int j = 0; j < 4; ++j) {
            acc[j] = __builtin_amdgcn_wmma_f32_16x16x32_bf16(
                false, a0.v, false, b[j].v, (short)0, acc[j], false, false);
        }
#pragma unroll
        for (int j = 0; j < 4; ++j) {
            acc[4 + j] = __builtin_amdgcn_wmma_f32_16x16x32_bf16(
                false, a1.v, false, b[j].v, (short)0, acc[4 + j], false, false);
        }
    };

    // ---- main K loop (full 32-wide steps, no branches inside) ----
    const int Kmain = K & ~31;
    for (int k0 = 0; k0 < Kmain; k0 += 32) {
        FragBF a0, a1, b[4];
        loadA(rowA0, k0, a0);
        loadA(rowA1, k0, a1);
        loadB(k0, b);
        mmaStep(a0, a1, b);
    }
    // ---- peeled K tail (f32-A path only; Bt is zero-padded to ldb) ----
    if (!A_BF16 && Kmain < K) {
        const int k0 = Kmain;
        FragBF a0, a1, b[4];
        auto loadATail = [&](int row, FragBF& a) {
            const float* ap = (const float*)Aptr + (size_t)row * lda;
#pragma unroll
            for (int i = 0; i < 8; ++i) {
                int k = k0 + kA + i;
                a.v[i] = (k < K) ? (__bf16)ap[k] : (__bf16)0.0f;
                int k2 = k + 16;
                a.v[8 + i] = (k2 < K) ? (__bf16)ap[k2] : (__bf16)0.0f;
            }
        };
        loadATail(rowA0, a0);
        loadATail(rowA1, a1);
        loadB(k0, b);
        mmaStep(a0, a1, b);
    }

    // C layout: VGPR r -> row base + r + (lane>>4)*8 ; col = n0 + j*16 + (lane&15)
#pragma unroll
    for (int g = 0; g < 2; ++g) {
#pragma unroll
        for (int j = 0; j < 4; ++j) {
            int c = n0 + j * 16 + (lane & 15);
#pragma unroll
            for (int r = 0; r < 8; ++r) {
                int row = m0 + g * 16 + r + (lane >> 4) * 8;
                if (row < M && c < Nout) {
                    float v = acc[g * 4 + j][r];
                    if (ACT == 1) v = (v > 20.f) ? v : log1pf(__expf(v));
                    C[(size_t)row * ldc + c] = v;
                }
            }
        }
    }
}

// ---------------------------------------------------------------------------
// Attention scalar scores: as1[n] = h_src[n,:] . a1s ; ad1[n] = h_dst[n,:] . a1d
// one wave per node
// ---------------------------------------------------------------------------
__global__ __launch_bounds__(256) void attn_scores(
    const float* __restrict__ hs, const float* __restrict__ hd,
    const float* __restrict__ a1s, const float* __restrict__ a1d,
    float* __restrict__ as1, float* __restrict__ ad1, int H)
{
    int node = blockIdx.x * 8 + (threadIdx.x >> 5);
    int lane = threadIdx.x & 31;
    float s = 0.f, d = 0.f;
    for (int k = lane; k < H; k += 32) {
        s += hs[(size_t)node * H + k] * a1s[k];
        d += hd[(size_t)node * H + k] * a1d[k];
    }
#pragma unroll
    for (int off = 16; off; off >>= 1) {
        s += __shfl_xor(s, off, 32);
        d += __shfl_xor(d, off, 32);
    }
    if (lane == 0) { as1[node] = s; ad1[node] = d; }
}

// order-preserving f32 <-> u32 mapping for atomicMax
__device__ __forceinline__ unsigned f2ord(float f) {
    union { float f; unsigned u; } c; c.f = f;
    return (c.u & 0x80000000u) ? ~c.u : (c.u | 0x80000000u);
}
__device__ __forceinline__ float ord2f(unsigned k) {
    union { unsigned u; float f; } c;
    c.u = (k & 0x80000000u) ? (k ^ 0x80000000u) : ~k;
    return c.f;
}

__global__ __launch_bounds__(256) void edge_max(
    const int* __restrict__ src, const int* __restrict__ dst,
    const float* __restrict__ as1, const float* __restrict__ ad1,
    float* __restrict__ ev, unsigned* __restrict__ mseg, int E)
{
    int e = blockIdx.x * 256 + threadIdx.x;
    if (e >= E) return;
    float v = as1[src[e]] + ad1[dst[e]];
    v = (v > 0.f) ? v : 0.2f * v;            // leaky_relu 0.2
    ev[e] = v;
    atomicMax(&mseg[dst[e]], f2ord(v));
}

__global__ __launch_bounds__(256) void edge_exp(
    const int* __restrict__ dst, float* __restrict__ ev,
    const unsigned* __restrict__ mseg, float* __restrict__ den, int E)
{
    int e = blockIdx.x * 256 + threadIdx.x;
    if (e >= E) return;
    float ex = __expf(ev[e] - ord2f(mseg[dst[e]]));
    ev[e] = ex;
    atomicAdd(&den[dst[e]], ex);
}

// one wave per edge: agg[dst] += feat[src] * alpha
__global__ __launch_bounds__(256) void edge_scatter(
    const int* __restrict__ src, const int* __restrict__ dst,
    const float* __restrict__ ev, const float* __restrict__ den,
    const float* __restrict__ feat, float* __restrict__ agg, int E, int H)
{
    int e = blockIdx.x * 8 + (threadIdx.x >> 5);
    if (e >= E) return;
    int s = src[e], d = dst[e];
    float alpha = ev[e] / (den[d] + 1e-16f);
    int lane = threadIdx.x & 31;
    const float4* fs = (const float4*)(feat + (size_t)s * H);
    float* ad = agg + (size_t)d * H;
    for (int j = lane; j < (H >> 2); j += 32) {
        float4 v = fs[j];
        atomicAdd(ad + j * 4 + 0, v.x * alpha);
        atomicAdd(ad + j * 4 + 1, v.y * alpha);
        atomicAdd(ad + j * 4 + 2, v.z * alpha);
        atomicAdd(ad + j * 4 + 3, v.w * alpha);
    }
}

// ---------------------------------------------------------------------------
// Fused small chain, one wave per node:
//   h1 = elu(agg) ; hidden = h1 @ W2s ; mean/log_var ; z ; g_src = z @ W2s^T
// ---------------------------------------------------------------------------
__global__ __launch_bounds__(256) void small_chain(
    const float* __restrict__ agg, const float* __restrict__ W2s,
    const float* __restrict__ Wm, const float* __restrict__ bm,
    const float* __restrict__ Wv, const float* __restrict__ bv,
    const float* __restrict__ eps, float* __restrict__ mean_o,
    float* __restrict__ lv_o, float* __restrict__ z_o,
    float* __restrict__ gsrc, int H, int L)
{
    __shared__ float h1s[8][512];
    __shared__ float hid[8][32];
    __shared__ float zs[8][32];
    const int wave = threadIdx.x >> 5;
    const int lane = threadIdx.x & 31;
    const int node = blockIdx.x * 8 + wave;

#pragma unroll
    for (int j = 0; j < 16; ++j) {
        int k = lane + j * 32;
        float v = agg[(size_t)node * H + k];
        h1s[wave][k] = (v > 0.f) ? v : expm1f(v);     // elu
    }
    __syncthreads();

    float h = 0.f;                                    // lane = output col c
    for (int k = 0; k < H; ++k) h += h1s[wave][k] * W2s[k * L + lane];
    hid[wave][lane] = h;
    __syncthreads();

    float mn = bm[lane], lv = bv[lane];
#pragma unroll
    for (int j = 0; j < 32; ++j) {
        float hj = hid[wave][j];
        mn += hj * Wm[j * L + lane];
        lv += hj * Wv[j * L + lane];
    }
    lv = fminf(10.f, fmaxf(-10.f, lv));
    float sd = sqrtf(__expf(0.5f * lv) + 1e-8f);
    float z = mn + eps[(size_t)node * L + lane] * sd;
    mean_o[(size_t)node * L + lane] = mn;
    lv_o[(size_t)node * L + lane] = lv;
    z_o[(size_t)node * L + lane] = z;
    zs[wave][lane] = z;
    __syncthreads();

#pragma unroll
    for (int j = 0; j < 16; ++j) {
        int k = lane + j * 32;
        float g = 0.f;
#pragma unroll
        for (int c = 0; c < 32; ++c) g += zs[wave][c] * W2s[k * L + c];
        gsrc[(size_t)node * H + k] = g;
    }
}

__global__ __launch_bounds__(256) void elu_to_bf16(
    const float* __restrict__ in, unsigned short* __restrict__ out, int n)
{
    int i = blockIdx.x * 256 + threadIdx.x;
    if (i >= n) return;
    float v = in[i];
    v = (v > 0.f) ? v : expm1f(v);
    out[i] = f32_to_bf16_rne(v);
}

__global__ __launch_bounds__(256) void theta_kernel(
    const float* __restrict__ lt, float* __restrict__ out, int n)
{
    int i = blockIdx.x * 256 + threadIdx.x;
    if (i < n) out[i] = __expf(lt[i]);
}

// ---------------------------------------------------------------------------
extern "C" void kernel_launch(void* const* d_in, const int* in_sizes, int n_in,
                              void* d_out, int out_size, void* d_ws, size_t ws_size,
                              hipStream_t stream)
{
    const int IN = 3000, HID = 512, LAT = 32, KP = 3008;
    const int N = in_sizes[0] / IN;
    const int E = in_sizes[1] / 2;

    const float* x   = (const float*)d_in[0];
    const int*   src = (const int*)d_in[1];           // edge_index[0]
    const int*   dst = src + E;                       // edge_index[1]
    const float* eps = (const float*)d_in[2];
    const float* W1s = (const float*)d_in[3];
    const float* W1d = (const float*)d_in[4];
    const float* a1s = (const float*)d_in[5];
    const float* a1d = (const float*)d_in[6];
    const float* W2s = (const float*)d_in[7];
    // d_in[8] = W2d unused by the reference
    const float* Wm  = (const float*)d_in[9];
    const float* bm  = (const float*)d_in[10];
    const float* Wv  = (const float*)d_in[11];
    const float* bv  = (const float*)d_in[12];
    const float* lt  = (const float*)d_in[13];

    float* out_mean = (float*)d_out;                          // N*LAT
    float* out_lv   = out_mean + (size_t)N * LAT;             // N*LAT
    float* out_mu   = out_lv + (size_t)N * LAT;               // N*IN
    float* out_th   = out_mu + (size_t)N * IN;                // IN
    float* out_z    = out_th + IN;                            // N*LAT

    // ---- workspace carve-up (256B aligned) ----
    char* ws = (char*)d_ws;
    size_t off = 0;
    auto carve = [&](size_t bytes) {
        char* p = ws + off;
        off += (bytes + 255) & ~(size_t)255;
        return p;
    };
    unsigned short* w1s_t = (unsigned short*)carve((size_t)HID * KP * 2);
    unsigned short* w1d_t = (unsigned short*)carve((size_t)HID * KP * 2);
    unsigned short* w1s_b = (unsigned short*)carve((size_t)IN * HID * 2);
    float* h_src = (float*)carve((size_t)N * HID * 4);
    float* h_dst = (float*)carve((size_t)N * HID * 4);   // later reused as g_src
    float* agg   = (float*)carve((size_t)N * HID * 4);   // reused for both aggregations
    unsigned short* h3b = (unsigned short*)carve((size_t)N * HID * 2);
    float* as1   = (float*)carve((size_t)N * 4);
    float* ad1   = (float*)carve((size_t)N * 4);
    unsigned* mseg = (unsigned*)carve((size_t)N * 4);
    float* den   = (float*)carve((size_t)N * 4);
    float* ev    = (float*)carve((size_t)E * 4);
    (void)ws_size;

    // ---- 1. weight prep ----
    {
        int n = HID * KP;
        w_transpose_bf16<<<(n + 255) / 256, 256, 0, stream>>>(W1s, w1s_t, IN, HID, KP);
        w_transpose_bf16<<<(n + 255) / 256, 256, 0, stream>>>(W1d, w1d_t, IN, HID, KP);
        int m = IN * HID;
        w_convert_bf16<<<(m + 255) / 256, 256, 0, stream>>>(W1s, w1s_b, m);
    }
    // ---- 2. zero accumulators ----
    {
        int n = N * HID;
        fill_u32<<<(n + 255) / 256, 256, 0, stream>>>((unsigned*)agg, 0u, n);
        fill_u32<<<(N + 255) / 256, 256, 0, stream>>>(mseg, 0u, N);
        fill_u32<<<(N + 255) / 256, 256, 0, stream>>>((unsigned*)den, 0u, N);
    }
    // ---- 3. encoder GAT1 projections (WMMA, A=f32 x) ----
    {
        dim3 grid(HID / 64, (N + 255) / 256);
        gemm_wmma<0, 0><<<grid, 256, 0, stream>>>(x, w1s_t, h_src, N, IN, HID, IN, KP, HID);
        gemm_wmma<0, 0><<<grid, 256, 0, stream>>>(x, w1d_t, h_dst, N, IN, HID, IN, KP, HID);
    }
    // ---- 4. attention scalars ----
    attn_scores<<<N / 8, 256, 0, stream>>>(h_src, h_dst, a1s, a1d, as1, ad1, HID);
    // ---- 5. edge softmax statistics (shared by encoder & decoder) ----
    edge_max<<<(E + 255) / 256, 256, 0, stream>>>(src, dst, as1, ad1, ev, mseg, E);
    edge_exp<<<(E + 255) / 256, 256, 0, stream>>>(dst, ev, mseg, den, E);
    // ---- 6. encoder aggregation: agg = softmax-weighted scatter of h_src ----
    edge_scatter<<<(E + 7) / 8, 256, 0, stream>>>(src, dst, ev, den, h_src, agg, E, HID);
    // ---- 7. fused latent chain (elu, hidden, mean/logvar/z, g_src) ----
    small_chain<<<N / 8, 256, 0, stream>>>(agg, W2s, Wm, bm, Wv, bv, eps,
                                           out_mean, out_lv, out_z, h_dst, HID, LAT);
    // ---- 8. decoder aggregation (same attention), reuse agg buffer ----
    {
        int n = N * HID;
        fill_u32<<<(n + 255) / 256, 256, 0, stream>>>((unsigned*)agg, 0u, n);
    }
    edge_scatter<<<(E + 7) / 8, 256, 0, stream>>>(src, dst, ev, den, h_dst, agg, E, HID);
    // ---- 9. h3 = elu(agg) -> bf16 ----
    {
        int n = N * HID;
        elu_to_bf16<<<(n + 255) / 256, 256, 0, stream>>>(agg, h3b, n);
    }
    // ---- 10. decoder reconstruction GEMM + softplus (WMMA, A=bf16) ----
    {
        dim3 grid((IN + 63) / 64, (N + 255) / 256);
        gemm_wmma<1, 1><<<grid, 256, 0, stream>>>(h3b, w1s_b, out_mu, N, HID, IN, HID, HID, IN);
    }
    // ---- 11. theta ----
    theta_kernel<<<(IN + 255) / 256, 256, 0, stream>>>(lt, out_th, IN);
}
